// NetEGNN_hid_ped_obs2_44822278701389
// MI455X (gfx1250) — compile-verified, matrix operand order checked by reading, not run on hardware
//
#include <hip/hip_runtime.h>
#include <math.h>

// ---------------------------------------------------------------------------
// Types
// ---------------------------------------------------------------------------
typedef _Float16 v16h __attribute__((ext_vector_type(16)));
typedef _Float16 h8   __attribute__((ext_vector_type(8)));
typedef float    v8f  __attribute__((ext_vector_type(8)));

#define HID 64
#define NPED 1024
#define NB   8
#define NOBS 256
#define NN   (NB * NPED)   // 8192 nodes

// ---------------------------------------------------------------------------
// WMMA helper (CDNA5: D = A(16x32 f16) * B(32x16 f16) + C(16x16 f32))
// ---------------------------------------------------------------------------
__device__ __forceinline__ v8f wmma_f16(v16h a, v16h b, v8f c) {
  return __builtin_amdgcn_wmma_f32_16x16x32_f16(false, a, false, b,
                                                (short)0, c, false, false);
}

// Fragment-element offset for 16-bit A layout (ISA 7.12.2):
//  lanes 0-15 hold K∈{0..7}∪{16..23}; lanes 16-31 hold K∈{8..15}∪{24..31}
__device__ __host__ __forceinline__ int a_off(int mtile, int KT, int kq, int m) {
  int ktile = kq >> 5;
  int kk    = kq & 31;
  int ln    = (m & 15) + (((kk & 15) >= 8) ? 16 : 0);
  int h     = (kk & 7) + ((kk >= 16) ? 8 : 0);
  return ((mtile * KT + ktile) * 32 + ln) * 16 + h;
}

__device__ __forceinline__ float silu(float x) {
  return x / (1.f + __expf(-x));
}

// ---------------------------------------------------------------------------
// Weight prep: convert fp32 (O x I, row-major, y[o]=sum_i x[i]*W[o,i]) into
// f16 WMMA B-fragments.  B element (k=i, n=o); mirrored lane/K mapping of A.
// Fragment array index == ((kt*NT+nt)*32+lane)*16+h, matching GEMM loads.
// ---------------------------------------------------------------------------
__global__ void prep_frag_kernel(const float* __restrict__ W,
                                 _Float16* __restrict__ frag,
                                 int O, int I, int KT, int NT) {
  int id = blockIdx.x * blockDim.x + threadIdx.x;
  int total = KT * NT * 512;
  if (id >= total) return;
  int kt  = id / (NT * 512);
  int rem = id % (NT * 512);
  int nt  = rem / 512;
  int q   = rem % 512;
  int ln  = q >> 4;
  int h   = q & 15;
  int k   = kt * 32 + (h & 7) + ((ln >= 16) ? 8 : 0) + ((h >= 8) ? 16 : 0);
  int o   = nt * 16 + (ln & 15);
  float v = (k < I && o < O) ? W[o * I + k] : 0.f;
  frag[id] = (_Float16)v;
}

// ---------------------------------------------------------------------------
// Init: h0 = [encV(|v|), encA(|a|), k_emb] (19) -> h_st = emb(64)
// ---------------------------------------------------------------------------
__global__ void init_kernel(const float* __restrict__ ped,
                            const float* __restrict__ kemb,
                            const float* __restrict__ encVw, const float* __restrict__ encVb,
                            const float* __restrict__ encAw, const float* __restrict__ encAb,
                            const float* __restrict__ embw,  const float* __restrict__ embb,
                            float* __restrict__ hout, _Float16* __restrict__ hout16) {
  int t = blockIdx.x * blockDim.x + threadIdx.x;
  if (t >= NN) return;
  const float* p = ped + (size_t)t * 6;
  float vn = sqrtf(p[2] * p[2] + p[3] * p[3]);
  float an = sqrtf(p[4] * p[4] + p[5] * p[5]);
  float h0[19];
#pragma unroll
  for (int o = 0; o < 8; ++o) {
    h0[o]     = encVw[o] * vn + encVb[o];
    h0[8 + o] = encAw[o] * an + encAb[o];
  }
#pragma unroll
  for (int c = 0; c < 3; ++c) h0[16 + c] = kemb[(size_t)t * 3 + c];
  for (int o = 0; o < 64; ++o) {
    float s = embb[o];
    const float* w = embw + o * 19;
#pragma unroll
    for (int i = 0; i < 19; ++i) s += w[i] * h0[i];
    hout[(size_t)t * 64 + o]   = s;
    hout16[(size_t)t * 64 + o] = (_Float16)s;
  }
}

// ---------------------------------------------------------------------------
// Fused graph-conv layer
// ---------------------------------------------------------------------------
struct ConvArgs {
  const float*    h_in;      // (B,N,64) fp32
  const _Float16* h_in16;    // (B,N,64) f16 (for neighbor gathers)
  float*          h_out;
  _Float16*       h_out16;
  float*          out_extra; // optional extra fp32 sink (d_out), may be null
  const float*    pf_in;     // (B,N,6) current kinematic state
  float*          pf_out;
  const float*    rel_src;   // ped: pf_in (stride 6); obs: obs_features (stride 2)
  int             rel_stride;
  int             rel_rows;  // ped: N; obs: M
  const int*      nidx;      // (B,N,KEDGES)
  const int*      nmask;
  const float*    ped_feat;  // original ped_features (for acce)
  const _Float16* frag_fe1;
  const _Float16* frag_fe2;
  const _Float16* frag_fx1;
  const float *fe_b1, *fe_b2;
  const float *fx_b1, *fx_w2, *fx_b2;
  const float *fa_w1, *fa_b1, *fa_w2, *fa_b2;
  const float *fh_w1, *fh_b1, *fh_w2, *fh_b2;
};

template <int KEDGES, int KT1, bool USE_NEIGH>
__global__ __launch_bounds__(128) void conv_kernel(ConvArgs a) {
  constexpr int MT = KEDGES / 16;

  __shared__ _Float16 sA1[MT * KT1 * 512]; // edge-feature A fragments / m fragments
  __shared__ _Float16 sA2[MT * 1024];      // hidden A fragments / plain y tile
  __shared__ float    sh_h[64];
  __shared__ _Float16 sh_hrep[64];
  __shared__ float    sh_mask[KEDGES];
  __shared__ int      sh_idx[KEDGES];
  __shared__ float    sh_rel[KEDGES * 2];
  __shared__ float    sh_dist[KEDGES];
  __shared__ float    sh_mi[64];
  __shared__ float    sh_red[128];
  __shared__ float    sh_t1[64];
  __shared__ float    sh_sc[4]; // aggx, aggy, neigh_cnt, fa

  const int tid  = threadIdx.x;
  const int lane = tid & 31;
  const int wave = tid >> 5;
  const int bn   = blockIdx.x;
  const int b    = bn >> 10;     // N == 1024
  const size_t nodebase = (size_t)bn * 64;

  // ---- phase 0: node state + edge metadata ------------------------------
  if (tid < 64) {
    float hv   = a.h_in[nodebase + tid];
    sh_h[tid]  = hv;
    sh_hrep[tid] = (_Float16)hv;
    sh_mi[tid] = 0.f;
  }
  if (tid < KEDGES) {
    size_t eb = (size_t)bn * KEDGES + tid;
    int   idx = a.nidx[eb];
    float mk  = (float)a.nmask[eb];
    sh_idx[tid]  = idx;
    sh_mask[tid] = mk;
    float sx = a.pf_in[(size_t)bn * 6 + 0];
    float sy = a.pf_in[(size_t)bn * 6 + 1];
    const float* rs = a.rel_src + ((size_t)b * a.rel_rows + idx) * a.rel_stride;
    float rx = (rs[0] - sx) * mk;
    float ry = (rs[1] - sy) * mk;
    sh_rel[tid * 2 + 0] = rx;
    sh_rel[tid * 2 + 1] = ry;
    sh_dist[tid] = sqrtf(rx * rx + ry * ry);
  }
  __syncthreads();

  // ---- phase 1: zero + assemble A (edge inputs) in fragment order -------
  for (int q = tid; q < MT * KT1 * 512; q += 128) sA1[q] = (_Float16)0.f;
  __syncthreads();

  for (int q = tid; q < KEDGES * 64; q += 128) { // replicated h_rep (k'=0..63)
    int e = q >> 6, k = q & 63;
    sA1[a_off(e >> 4, KT1, k, e & 15)] = sh_hrep[k];
  }
  if (USE_NEIGH) { // gathered, masked h_neigh (k'=64..127); 2 threads/edge
    int e  = tid >> 1;
    int kb = (tid & 1) * 32;
    const h8* src = (const h8*)(a.h_in16 + ((size_t)b * NPED + sh_idx[e]) * 64 + kb);
    _Float16 mh = (_Float16)sh_mask[e];
#pragma unroll
    for (int c = 0; c < 4; ++c) {
      h8 v = src[c];
#pragma unroll
      for (int j = 0; j < 8; ++j) {
        int kq = 64 + kb + c * 8 + j;
        sA1[a_off(e >> 4, KT1, kq, e & 15)] = v[j] * mh;
      }
    }
  }
  if (tid < KEDGES) { // distance scalar
    int kq = USE_NEIGH ? 128 : 64;
    sA1[a_off(tid >> 4, KT1, kq, tid & 15)] = (_Float16)sh_dist[tid];
  }
  __syncthreads();

  const int nf_lo = lane & 15;
  const int mbase = (lane >> 4) * 8;

  // ---- fe layer 1: [KEDGES x KT1*32] x [.. x 64] -------------------------
  if (wave < MT) {
    v8f acc[4];
#pragma unroll
    for (int nt = 0; nt < 4; ++nt)
#pragma unroll
      for (int j = 0; j < 8; ++j) acc[nt][j] = 0.f;
#pragma unroll
    for (int kt = 0; kt < KT1; ++kt) {
      v16h av = *(const v16h*)&sA1[((wave * KT1 + kt) * 32 + lane) * 16];
#pragma unroll
      for (int nt = 0; nt < 4; ++nt) {
        v16h bv = *(const v16h*)(a.frag_fe1 + ((kt * 4 + nt) * 32 + lane) * 16);
        acc[nt] = wmma_f16(av, bv, acc[nt]);
      }
    }
#pragma unroll
    for (int nt = 0; nt < 4; ++nt) {
      float bias = a.fe_b1[nt * 16 + nf_lo];
#pragma unroll
      for (int r = 0; r < 8; ++r) {
        float s = silu(acc[nt][r] + bias);
        sA2[a_off(wave, 2, nt * 16 + nf_lo, mbase + r)] = (_Float16)s;
      }
    }
  }
  __syncthreads();

  // ---- fe layer 2 -> m (masked); accumulate m_i --------------------------
  if (wave < MT) {
    v8f acc[4];
#pragma unroll
    for (int nt = 0; nt < 4; ++nt)
#pragma unroll
      for (int j = 0; j < 8; ++j) acc[nt][j] = 0.f;
#pragma unroll
    for (int kt = 0; kt < 2; ++kt) {
      v16h av = *(const v16h*)&sA2[((wave * 2 + kt) * 32 + lane) * 16];
#pragma unroll
      for (int nt = 0; nt < 4; ++nt) {
        v16h bv = *(const v16h*)(a.frag_fe2 + ((kt * 4 + nt) * 32 + lane) * 16);
        acc[nt] = wmma_f16(av, bv, acc[nt]);
      }
    }
#pragma unroll
    for (int nt = 0; nt < 4; ++nt) {
      float bias = a.fe_b2[nt * 16 + nf_lo];
      float part = 0.f;
#pragma unroll
      for (int r = 0; r < 8; ++r) {
        int m = mbase + r;
        float s = silu(acc[nt][r] + bias) * sh_mask[wave * 16 + m];
        part += s;
        sA1[a_off(wave, KT1, nt * 16 + nf_lo, m)] = (_Float16)s; // reuse as m-frag
      }
      atomicAdd(&sh_mi[nt * 16 + nf_lo], part);
    }
  }
  __syncthreads();

  // ---- fx layer 1 --------------------------------------------------------
  if (wave < MT) {
    v8f acc[4];
#pragma unroll
    for (int nt = 0; nt < 4; ++nt)
#pragma unroll
      for (int j = 0; j < 8; ++j) acc[nt][j] = 0.f;
#pragma unroll
    for (int kt = 0; kt < 2; ++kt) {
      v16h av = *(const v16h*)&sA1[((wave * KT1 + kt) * 32 + lane) * 16];
#pragma unroll
      for (int nt = 0; nt < 4; ++nt) {
        v16h bv = *(const v16h*)(a.frag_fx1 + ((kt * 4 + nt) * 32 + lane) * 16);
        acc[nt] = wmma_f16(av, bv, acc[nt]);
      }
    }
#pragma unroll
    for (int nt = 0; nt < 4; ++nt) {
      float bias = a.fx_b1[nt * 16 + nf_lo];
#pragma unroll
      for (int r = 0; r < 8; ++r) {
        int m = mbase + r;
        sA2[wave * 1024 + m * 64 + nt * 16 + nf_lo] = (_Float16)silu(acc[nt][r] + bias);
      }
    }
  }
  __syncthreads();

  // ---- fx layer 2 (dot) + aggregation partials ---------------------------
  if (tid < KEDGES) {
    const _Float16* yrow = &sA2[(tid >> 4) * 1024 + (tid & 15) * 64];
    float s = a.fx_b2[0];
#pragma unroll
    for (int k = 0; k < 64; ++k) s += (float)yrow[k] * a.fx_w2[k];
    sh_red[tid]          = sh_rel[tid * 2 + 0] * s;
    sh_red[KEDGES + tid] = sh_rel[tid * 2 + 1] * s;
  }
  __syncthreads();
  if (tid == 0) { float s = 0.f; for (int e = 0; e < KEDGES; ++e) s += sh_red[e];          sh_sc[0] = s; }
  if (tid == 1) { float s = 0.f; for (int e = 0; e < KEDGES; ++e) s += sh_red[KEDGES + e]; sh_sc[1] = s; }
  if (tid == 2) { float s = 0.f; for (int e = 0; e < KEDGES; ++e) s += sh_mask[e];         sh_sc[2] = s; }
  __syncthreads();

  // ---- fa: node MLP -> scalar -------------------------------------------
  if (tid < 64) {
    float t = a.fa_b1[tid];
    const float* w = a.fa_w1 + tid * 64;
#pragma unroll 8
    for (int i = 0; i < 64; ++i) t += sh_h[i] * w[i];
    sh_red[tid] = silu(t) * a.fa_w2[tid];
  }
  __syncthreads();
  if (tid == 0) {
    float s = a.fa_b2[0];
    for (int i = 0; i < 64; ++i) s += sh_red[i];
    sh_sc[3] = s;
  }
  // ---- fh: node MLP on [h, m_i] (runs alongside fa reduce) ---------------
  if (tid < 64) {
    float t = a.fh_b1[tid];
    const float* w = a.fh_w1 + tid * 128;
#pragma unroll 8
    for (int i = 0; i < 64; ++i) t += sh_h[i] * w[i];
#pragma unroll 8
    for (int i = 0; i < 64; ++i) t += sh_mi[i] * w[64 + i];
    sh_t1[tid] = silu(t);
  }
  __syncthreads();
  if (tid < 64) {
    float s = a.fh_b2[tid];
    const float* w = a.fh_w2 + tid * 64;
#pragma unroll 8
    for (int i = 0; i < 64; ++i) s += sh_t1[i] * w[i];
    float hn = sh_h[tid] + s;
    a.h_out[nodebase + tid]   = hn;
    a.h_out16[nodebase + tid] = (_Float16)hn;
    if (a.out_extra) a.out_extra[nodebase + tid] = hn;
  }
  // ---- kinematic update --------------------------------------------------
  if (tid == 0) {
    float inv = 1.f / (sh_sc[2] + 1e-6f);
    float fa  = sh_sc[3];
    const float* pf = a.pf_in + (size_t)bn * 6;
    float pout[6];
#pragma unroll
    for (int c = 0; c < 2; ++c) {
      float acce = a.ped_feat[(size_t)bn * 6 + 4 + c];
      float an = fa * acce + sh_sc[c] * inv;
      float vn = pf[2 + c] + an;
      float xn = pf[c] + vn;
      pout[c] = xn; pout[2 + c] = vn; pout[4 + c] = an;
    }
#pragma unroll
    for (int c = 0; c < 6; ++c) a.pf_out[(size_t)bn * 6 + c] = pout[c];
  }
}

// ---------------------------------------------------------------------------
// Host launcher
// ---------------------------------------------------------------------------
extern "C" void kernel_launch(void* const* d_in, const int* in_sizes, int n_in,
                              void* d_out, int out_size, void* d_ws, size_t ws_size,
                              hipStream_t stream) {
  const float* ped    = (const float*)d_in[0];
  const float* kemb   = (const float*)d_in[1];
  const float* obs    = (const float*)d_in[2];
  const int*   nmask  = (const int*)d_in[3];
  const int*   nidx   = (const int*)d_in[4];
  const int*   nmaskO = (const int*)d_in[5];
  const int*   nidxO  = (const int*)d_in[6];
  const float* encVw  = (const float*)d_in[7];
  const float* encVb  = (const float*)d_in[8];
  const float* encAw  = (const float*)d_in[9];
  const float* encAb  = (const float*)d_in[10];
  const float* embw   = (const float*)d_in[11];
  const float* embb   = (const float*)d_in[12];
  // block params: fe_w1,fe_b1,fe_w2,fe_b2,fx_w1,fx_b1,fx_w2,fx_b2,
  //               fa_w1,fa_b1,fa_w2,fa_b2,fh_w1,fh_b1,fh_w2,fh_b2
  const float* G[16]; const float* O[16];
  for (int i = 0; i < 16; ++i) G[i] = (const float*)d_in[13 + i];
  for (int i = 0; i < 16; ++i) O[i] = (const float*)d_in[29 + i];

  // workspace carve-up
  char* w = (char*)d_ws; size_t off = 0;
  auto take = [&](size_t bytes) -> char* {
    char* p = w + off; off += (bytes + 255) & ~(size_t)255; return p;
  };
  float*     hst0  = (float*)take((size_t)NN * 64 * 4);
  float*     hst1  = (float*)take((size_t)NN * 64 * 4);
  float*     hst2  = (float*)take((size_t)NN * 64 * 4);
  _Float16*  h16_0 = (_Float16*)take((size_t)NN * 64 * 2);
  _Float16*  h16_1 = (_Float16*)take((size_t)NN * 64 * 2);
  _Float16*  h16_2 = (_Float16*)take((size_t)NN * 64 * 2);
  float*     ho1   = (float*)take((size_t)NN * 64 * 4);
  float*     ho2   = (float*)take((size_t)NN * 64 * 4);
  _Float16*  ho16_1 = (_Float16*)take((size_t)NN * 64 * 2);
  _Float16*  ho16_2 = (_Float16*)take((size_t)NN * 64 * 2);
  float*     pfA  = (float*)take((size_t)NN * 6 * 4);
  float*     pfB  = (float*)take((size_t)NN * 6 * 4);
  float*     pfoA = (float*)take((size_t)NN * 6 * 4);
  float*     pfoB = (float*)take((size_t)NN * 6 * 4);
  _Float16*  fragG = (_Float16*)take((size_t)3 * 18432 * 2);
  _Float16*  fragO = (_Float16*)take((size_t)3 * 14336 * 2);

  // --- weight prep (f16 B-fragments) ---
  auto prep = [&](const float* W, _Float16* F, int Odim, int Idim, int KT) {
    int total = KT * 4 * 512;
    prep_frag_kernel<<<(total + 255) / 256, 256, 0, stream>>>(W, F, Odim, Idim, KT, 4);
  };
  for (int l = 0; l < 3; ++l) {
    prep(G[0] + (size_t)l * 64 * 129, fragG + (size_t)l * 18432,          64, 129, 5);
    prep(G[2] + (size_t)l * 64 * 64,  fragG + (size_t)l * 18432 + 10240,  64, 64,  2);
    prep(G[4] + (size_t)l * 64 * 64,  fragG + (size_t)l * 18432 + 14336,  64, 64,  2);
    prep(O[0] + (size_t)l * 64 * 65,  fragO + (size_t)l * 14336,          64, 65,  3);
    prep(O[2] + (size_t)l * 64 * 64,  fragO + (size_t)l * 14336 + 6144,   64, 64,  2);
    prep(O[4] + (size_t)l * 64 * 64,  fragO + (size_t)l * 14336 + 10240,  64, 64,  2);
  }

  // --- initial embedding ---
  init_kernel<<<(NN + 255) / 256, 256, 0, stream>>>(ped, kemb, encVw, encVb,
                                                    encAw, encAb, embw, embb,
                                                    hst0, h16_0);

  auto fill_common = [&](ConvArgs& ar, const float* P[16], int l) {
    ar.fe_b1 = P[1]  + (size_t)l * 64;
    ar.fe_b2 = P[3]  + (size_t)l * 64;
    ar.fx_b1 = P[5]  + (size_t)l * 64;
    ar.fx_w2 = P[6]  + (size_t)l * 64;
    ar.fx_b2 = P[7]  + (size_t)l;
    ar.fa_w1 = P[8]  + (size_t)l * 64 * 64;
    ar.fa_b1 = P[9]  + (size_t)l * 64;
    ar.fa_w2 = P[10] + (size_t)l * 64;
    ar.fa_b2 = P[11] + (size_t)l;
    ar.fh_w1 = P[12] + (size_t)l * 64 * 128;
    ar.fh_b1 = P[13] + (size_t)l * 64;
    ar.fh_w2 = P[14] + (size_t)l * 64 * 64;
    ar.fh_b2 = P[15] + (size_t)l * 64;
    ar.ped_feat = ped;
  };

  // --- pedestrian block (3 layers) ---
  {
    const float*    hin[3]    = {hst0, hst1, hst2};
    float*          hout[3]   = {hst1, hst2, hst1};
    const _Float16* hin16[3]  = {h16_0, h16_1, h16_2};
    _Float16*       hout16[3] = {h16_1, h16_2, h16_1};
    const float*    pfin[3]   = {ped, pfB, pfA};
    float*          pfout[3]  = {pfB, pfA, pfB};
    for (int l = 0; l < 3; ++l) {
      ConvArgs ar{};
      ar.h_in = hin[l];   ar.h_in16 = hin16[l];
      ar.h_out = hout[l]; ar.h_out16 = hout16[l];
      ar.out_extra = (l == 2) ? (float*)d_out : nullptr;
      ar.pf_in = pfin[l]; ar.pf_out = pfout[l];
      ar.rel_src = pfin[l]; ar.rel_stride = 6; ar.rel_rows = NPED;
      ar.nidx = nidx; ar.nmask = nmask;
      ar.frag_fe1 = fragG + (size_t)l * 18432;
      ar.frag_fe2 = fragG + (size_t)l * 18432 + 10240;
      ar.frag_fx1 = fragG + (size_t)l * 18432 + 14336;
      fill_common(ar, G, l);
      conv_kernel<64, 5, true><<<NN, 128, 0, stream>>>(ar);
    }
  }

  // --- obstacle block (3 layers), starts from the initial h_st ---
  {
    const float*    hin[3]    = {hst0, ho1, ho2};
    float*          hout[3]   = {ho1, ho2, ho1};
    const _Float16* hin16[3]  = {h16_0, ho16_1, ho16_2};
    _Float16*       hout16[3] = {ho16_1, ho16_2, ho16_1};
    const float*    pfin[3]   = {ped, pfoB, pfoA};
    float*          pfout[3]  = {pfoB, pfoA, pfoB};
    for (int l = 0; l < 3; ++l) {
      ConvArgs ar{};
      ar.h_in = hin[l];   ar.h_in16 = hin16[l];
      ar.h_out = hout[l]; ar.h_out16 = hout16[l];
      ar.out_extra = (l == 2) ? ((float*)d_out + (size_t)NN * 64) : nullptr;
      ar.pf_in = pfin[l]; ar.pf_out = pfout[l];
      ar.rel_src = obs; ar.rel_stride = 2; ar.rel_rows = NOBS;
      ar.nidx = nidxO; ar.nmask = nmaskO;
      ar.frag_fe1 = fragO + (size_t)l * 14336;
      ar.frag_fe2 = fragO + (size_t)l * 14336 + 6144;
      ar.frag_fx1 = fragO + (size_t)l * 14336 + 10240;
      fill_common(ar, O, l);
      conv_kernel<16, 3, false><<<NN, 128, 0, stream>>>(ar);
    }
  }

  (void)in_sizes; (void)n_in; (void)out_size; (void)ws_size;
}
